// MPNN_33638183862786
// MI455X (gfx1250) — matrix operand order repaired and checked by reading, not compiled, not run on hardware
//
#include <hip/hip_runtime.h>
#include <math.h>

typedef __bf16 bf16_t;
typedef __attribute__((ext_vector_type(16))) __bf16 bf16x16;
typedef __attribute__((ext_vector_type(8)))  __bf16 bf16x8;
typedef __attribute__((ext_vector_type(8)))  float  f32x8;

#define B_    8
#define N_    256
#define D_    128
#define MID_  128
#define FEAT_ 256
#define ROWS_ (B_ * N_)        // 2048
#define ESTR  136              // padded LDS row stride (bf16 elems) to break bank conflicts
#define NEG_BIG (-3.402823466e38f)

// Assemble a 16-element bf16 WMMA fragment from two 16-byte chunks.
__device__ __forceinline__ bf16x16 load_frag16(const bf16_t* p0, const bf16_t* p1) {
  bf16x8 lo = *(const bf16x8*)p0;
  bf16x8 hi = *(const bf16x8*)p1;
  bf16x16 r;
#pragma unroll
  for (int t = 0; t < 8; ++t) { r[t] = lo[t]; r[t + 8] = hi[t]; }
  return r;
}

// ---------------- prep kernels ----------------

// featbf[b*N+n][0:256] = concat(n_features, hidden), f32 -> bf16
__global__ __launch_bounds__(256) void kprep_feat(const float* __restrict__ nfeat,
                                                  const float* __restrict__ hidden,
                                                  bf16_t* __restrict__ featbf) {
  size_t idx = (size_t)blockIdx.x * 256 + threadIdx.x;   // over 2048*256
  int d = (int)(idx & 255);
  size_t row = idx >> 8;
  float v = (d < D_) ? nfeat[row * D_ + d] : hidden[row * D_ + (d - D_)];
  featbf[idx] = (bf16_t)v;
}

// WT[m][k] = W[k][m], f32 -> bf16 (K-contiguous per output column)
__global__ __launch_bounds__(256) void kprep_wT(const float* __restrict__ W, int K,
                                                bf16_t* __restrict__ WT) {
  int idx = blockIdx.x * 256 + threadIdx.x;              // over K*MID
  int m = idx / K, k = idx - m * K;
  WT[idx] = (bf16_t)W[(size_t)k * MID_ + m];
}

// Gc[b][m] = g_features[b] @ Wg + bg[m] + be[m] + b1[m]
__global__ __launch_bounds__(128) void kprep_gc(const float* __restrict__ g,
                                                const float* __restrict__ Wg,
                                                const float* __restrict__ bg,
                                                const float* __restrict__ be,
                                                const float* __restrict__ b1,
                                                float* __restrict__ Gc) {
  int b = blockIdx.x, m = threadIdx.x;
  float s = bg[m] + be[m] + b1[m];
#pragma unroll 4
  for (int d = 0; d < D_; ++d) s += g[b * D_ + d] * Wg[d * MID_ + m];
  Gc[b * MID_ + m] = s;
}

__global__ __launch_bounds__(128) void kprep_bias(const float* __restrict__ bo1,
                                                  const float* __restrict__ bo2,
                                                  float* __restrict__ bo12) {
  int m = threadIdx.x;
  bo12[m] = bo1[m] + bo2[m];
}

// ---------------- generic wave-tiled bf16 WMMA GEMM ----------------
// out[r][m] = X[r][:] @ WT[m][:]  (+ addM[m]) (+ addBM[b][m]) (+ addRM[r][m])
// R = 2048 rows, M = 128 cols. Block = 256 thr (8 waves); wave w -> 16-row strip,
// loops over all 8 col-tiles. K templated so A fragments stay in registers.
template <int K>
__global__ __launch_bounds__(256) void gemm_bf16(const bf16_t* __restrict__ X,
                                                 const bf16_t* __restrict__ WT,
                                                 const float* __restrict__ addM,
                                                 const float* __restrict__ addBM,
                                                 const float* __restrict__ addRM,
                                                 float* __restrict__ out) {
  constexpr int KS = K / 32;
  const int tid = threadIdx.x;
  const int wave = tid >> 5, lane = tid & 31;
  const bool hi = lane >= 16;
  const int colLane = lane & 15;
  const int r0 = blockIdx.x * 128 + wave * 16;           // row strip base
  const int arow = r0 + colLane;                         // A row held by this lane

  bf16x16 afr[KS];
#pragma unroll
  for (int ks = 0; ks < KS; ++ks) {
    const bf16_t* ab = X + (size_t)arow * K + ks * 32 + (hi ? 8 : 0);
    afr[ks] = load_frag16(ab, ab + 16);
  }

#pragma unroll
  for (int ct = 0; ct < 8; ++ct) {
    const int mcol = ct * 16 + colLane;
    f32x8 acc = {};
#pragma unroll
    for (int ks = 0; ks < KS; ++ks) {
      const bf16_t* bb = WT + (size_t)mcol * K + ks * 32 + (hi ? 8 : 0);
      bf16x16 bfr = load_frag16(bb, bb + 16);
      acc = __builtin_amdgcn_wmma_f32_16x16x32_bf16(false, afr[ks], false, bfr,
                                                    (short)0, acc, false, false);
    }
    float addc = addM ? addM[mcol] : 0.0f;
#pragma unroll
    for (int rr = 0; rr < 8; ++rr) {
      int row = r0 + (hi ? 8 : 0) + rr;
      float v = acc[rr] + addc;
      if (addBM) v += addBM[(row >> 8) * MID_ + mcol];
      if (addRM) v += addRM[(size_t)row * MID_ + mcol];
      out[(size_t)row * MID_ + mcol] = v;
    }
  }
}

// ---------------- main fused kernel ----------------
// One block per (b, j). msgs[b,j,m] = max_i adj[b,i,j] * (E[b,i,j,:]@We + C1[b,j,m] + C2[b,i,m])
// 8 waves: wave w owns m-tile w (cols 16w..16w+15) -> wave-private max reduction.
__global__ __launch_bounds__(256) void mpnn_main(const float* __restrict__ efeat,
                                                 const float* __restrict__ adj,
                                                 const bf16_t* __restrict__ WeT,
                                                 const float* __restrict__ C1,
                                                 const float* __restrict__ C2,
                                                 bf16_t* __restrict__ msgsbf) {
  __shared__ bf16_t elds[64 * ESTR];
  __shared__ float adjlds[64];

  const int b = blockIdx.y, j = blockIdx.x;
  const int tid = threadIdx.x;
  const int wave = tid >> 5, lane = tid & 31;
  const bool hi = lane >= 16;
  const int colLane = lane & 15;
  const int mcol = wave * 16 + colLane;

  // B fragments (We column tile): constant across all i -> keep in registers.
  bf16x16 bfr[4];
#pragma unroll
  for (int ks = 0; ks < 4; ++ks) {
    const bf16_t* bb = WeT + (size_t)mcol * D_ + ks * 32 + (hi ? 8 : 0);
    bfr[ks] = load_frag16(bb, bb + 16);
  }

  const float c1v = C1[((size_t)b * N_ + j) * MID_ + mcol];
  float runMax = NEG_BIG;

  const size_t ebase = ((size_t)b * N_ * N_ + j) * D_;   // + i * N_*D_
  const int rowL = tid >> 2;                              // 0..63 (staging row)
  const int seg  = (tid & 3) * 32;                        // 32-float segment

  for (int ic = 0; ic < 4; ++ic) {                        // 4 chunks of 64 i-rows
    __syncthreads();
    {
      const int i = ic * 64 + rowL;
      const float* src = efeat + ebase + (size_t)i * (N_ * D_) + seg;
      bf16_t tmp[32];
#pragma unroll
      for (int q = 0; q < 8; ++q) {
        float4 f = *(const float4*)(src + q * 4);
        tmp[q * 4 + 0] = (bf16_t)f.x; tmp[q * 4 + 1] = (bf16_t)f.y;
        tmp[q * 4 + 2] = (bf16_t)f.z; tmp[q * 4 + 3] = (bf16_t)f.w;
      }
      bf16_t* dst = elds + rowL * ESTR + seg;
#pragma unroll
      for (int q = 0; q < 4; ++q) *(bf16x8*)(dst + q * 8) = *(const bf16x8*)(tmp + q * 8);
      if (tid < 64) adjlds[tid] = adj[((size_t)b * N_ + (ic * 64 + tid)) * N_ + j];
    }
    __syncthreads();

#pragma unroll
    for (int it = 0; it < 4; ++it) {                      // 16-row i-tiles in chunk
      const int rb = it * 16;
      const bf16_t* abase = elds + (rb + colLane) * ESTR + (hi ? 8 : 0);
      f32x8 acc = {};
#pragma unroll
      for (int ks = 0; ks < 4; ++ks) {
        bf16x16 a = load_frag16(abase + ks * 32, abase + ks * 32 + 16);
        acc = __builtin_amdgcn_wmma_f32_16x16x32_bf16(false, a, false, bfr[ks],
                                                      (short)0, acc, false, false);
      }
#pragma unroll
      for (int rr = 0; rr < 8; ++rr) {
        const int il = rb + (hi ? 8 : 0) + rr;            // local i in chunk
        const int i = ic * 64 + il;
        float v = acc[rr] + c1v + C2[((size_t)b * N_ + i) * MID_ + mcol];
        runMax = fmaxf(runMax, v * adjlds[il]);           // adj is exactly 0.0 or 1.0
      }
    }
  }

  float other = __shfl_xor(runMax, 16, 32);
  runMax = fmaxf(runMax, other);
  if (lane < 16) msgsbf[((size_t)b * N_ + j) * MID_ + mcol] = (bf16_t)runMax;
}

// ---------------- launch ----------------
extern "C" void kernel_launch(void* const* d_in, const int* in_sizes, int n_in,
                              void* d_out, int out_size, void* d_ws, size_t ws_size,
                              hipStream_t stream) {
  const float* hidden = (const float*)d_in[0];
  const float* nfeat  = (const float*)d_in[1];
  const float* efeat  = (const float*)d_in[2];
  const float* gfeat  = (const float*)d_in[3];
  const float* adj    = (const float*)d_in[4];
  const float* W1  = (const float*)d_in[5];
  const float* b1  = (const float*)d_in[6];
  const float* W2  = (const float*)d_in[7];
  const float* b2  = (const float*)d_in[8];
  const float* We  = (const float*)d_in[9];
  const float* be  = (const float*)d_in[10];
  const float* Wg  = (const float*)d_in[11];
  const float* bg  = (const float*)d_in[12];
  const float* Wo1 = (const float*)d_in[13];
  const float* bo1 = (const float*)d_in[14];
  const float* Wo2 = (const float*)d_in[15];
  const float* bo2 = (const float*)d_in[16];
  float* out = (float*)d_out;

  char* w = (char*)d_ws;
  auto alloc = [&](size_t bytes) -> char* {
    char* p = w;
    w += (bytes + 255) & ~(size_t)255;
    return p;
  };
  bf16_t* featbf = (bf16_t*)alloc((size_t)ROWS_ * FEAT_ * 2);
  bf16_t* W1T    = (bf16_t*)alloc((size_t)FEAT_ * MID_ * 2);
  bf16_t* W2T    = (bf16_t*)alloc((size_t)FEAT_ * MID_ * 2);
  bf16_t* Wo1T   = (bf16_t*)alloc((size_t)FEAT_ * MID_ * 2);
  bf16_t* WeT    = (bf16_t*)alloc((size_t)D_ * MID_ * 2);
  bf16_t* Wo2T   = (bf16_t*)alloc((size_t)MID_ * MID_ * 2);
  bf16_t* msgsbf = (bf16_t*)alloc((size_t)ROWS_ * MID_ * 2);
  float*  Gc     = (float*)alloc((size_t)B_ * MID_ * 4);
  float*  bo12   = (float*)alloc((size_t)MID_ * 4);
  float*  C1     = (float*)alloc((size_t)ROWS_ * MID_ * 4);
  float*  C2     = (float*)alloc((size_t)ROWS_ * MID_ * 4);
  float*  lin1   = (float*)alloc((size_t)ROWS_ * MID_ * 4);

  // prep
  kprep_feat<<<ROWS_, 256, 0, stream>>>(nfeat, hidden, featbf);
  kprep_wT<<<(FEAT_ * MID_) / 256, 256, 0, stream>>>(W1, FEAT_, W1T);
  kprep_wT<<<(FEAT_ * MID_) / 256, 256, 0, stream>>>(W2, FEAT_, W2T);
  kprep_wT<<<(FEAT_ * MID_) / 256, 256, 0, stream>>>(Wo1, FEAT_, Wo1T);
  kprep_wT<<<(D_ * MID_) / 256, 256, 0, stream>>>(We, D_, WeT);
  kprep_wT<<<(MID_ * MID_) / 256, 256, 0, stream>>>(Wo2, MID_, Wo2T);
  kprep_gc<<<B_, 128, 0, stream>>>(gfeat, Wg, bg, be, b1, Gc);
  kprep_bias<<<1, 128, 0, stream>>>(bo1, bo2, bo12);

  // small GEMMs: C1, C2, lin1
  gemm_bf16<FEAT_><<<ROWS_ / 128, 256, 0, stream>>>(featbf, W1T, nullptr, Gc, nullptr, C1);
  gemm_bf16<FEAT_><<<ROWS_ / 128, 256, 0, stream>>>(featbf, W2T, b2, nullptr, nullptr, C2);
  gemm_bf16<FEAT_><<<ROWS_ / 128, 256, 0, stream>>>(featbf, Wo1T, bo12, nullptr, nullptr, lin1);

  // fused big GEMM + broadcast-add + masked max
  mpnn_main<<<dim3(N_, B_), 256, 0, stream>>>(efeat, adj, WeT, C1, C2, msgsbf);

  // out = lin1 + msgs @ Wo2
  gemm_bf16<MID_><<<ROWS_ / 128, 256, 0, stream>>>(msgsbf, Wo2T, nullptr, nullptr, lin1, out);
}
